// ProceduralLanguageModel_50199577756276
// MI455X (gfx1250) — matrix operand order repaired and checked by reading, not compiled, not run on hardware
//
#include <hip/hip_runtime.h>
#include <math.h>

typedef __bf16 bf16;
typedef bf16 bf16x16 __attribute__((ext_vector_type(16)));
typedef float f32x8 __attribute__((ext_vector_type(8)));
typedef unsigned int u32;
typedef u32 u32x4 __attribute__((ext_vector_type(4)));
typedef u32 u32x8 __attribute__((ext_vector_type(8)));
typedef unsigned short u16;
typedef int v4i __attribute__((vector_size(16)));  // matches builtin's V4i

#define B_ 8
#define S_ 2048
#define D_ 1024
#define H_ 1024
#define V_ 50257

// ---------------------------------------------------------------- CDNA5 async copy

#if __has_builtin(__builtin_amdgcn_global_load_async_to_lds_b128)
#define HAVE_ASYNC_LDS 1
#else
#define HAVE_ASYNC_LDS 0
#endif

// 16-byte global -> LDS copy. Async path uses the CDNA5 async-DMA path
// (GLOBAL_LOAD_ASYNC_TO_LDS_B128, tracked by ASYNCcnt); fallback is a
// synchronous VGPR round-trip.
__device__ __forceinline__ void async_copy16(const void* g, void* l) {
#if HAVE_ASYNC_LDS
  __builtin_amdgcn_global_load_async_to_lds_b128(
      (__attribute__((address_space(1))) v4i*)(g),
      (__attribute__((address_space(3))) v4i*)(l), 0, 0);
#else
  *(u32x4*)l = *(const u32x4*)g;
#endif
}

template <int N>
__device__ __forceinline__ void wait_async() {
#if HAVE_ASYNC_LDS
#if __has_builtin(__builtin_amdgcn_s_wait_asynccnt)
  __builtin_amdgcn_s_wait_asynccnt(N);
#else
  asm volatile("s_wait_asynccnt %0" ::"i"(N) : "memory");
#endif
#endif
}

// ---------------------------------------------------------------- helpers

__device__ __forceinline__ u16 f2bf(float f) {
  u32 u = __builtin_bit_cast(u32, f);
  u32 r = u + 0x7FFFu + ((u >> 16) & 1u);
  return (u16)(r >> 16);
}

__device__ __forceinline__ f32x8 zero8() {
  f32x8 z;
#pragma unroll
  for (int i = 0; i < 8; ++i) z[i] = 0.f;
  return z;
}

__device__ __forceinline__ bf16x16 frag_combine(u32x4 lo, u32x4 hi) {
  u32x8 t;
#pragma unroll
  for (int i = 0; i < 4; ++i) { t[i] = lo[i]; t[i + 4] = hi[i]; }
  return __builtin_bit_cast(bf16x16, t);
}

// A fragment (16 rows x 32 K) from a row-major bf16 LDS tile.
// ISA layout: lanes 0-15 = rows 0-15 holding K 0..7 & 16..23,
//             lanes 16-31 = rows 0-15 holding K 8..15 & 24..31.
__device__ __forceinline__ bf16x16 a_frag(const u16* tile, int pitch, int colBase, int lane) {
  int half = lane >> 4, r = lane & 15;
  const char* base = (const char*)(tile + r * pitch + colBase);
  u32x4 lo = *(const u32x4*)(base + half * 16);
  u32x4 hi = *(const u32x4*)(base + 32 + half * 16);
  return frag_combine(lo, hi);
}

// B fragment (32 K x 16 N) from an LDS tile stored as Bsrc[n][k] row-major.
// ISA layout: lanes 0-15 hold K 0..7 & 16..23 for column N=lane; lanes 16-31 hold K 8..15 & 24..31.
__device__ __forceinline__ bf16x16 b_frag(const u16* tile, int pitch, int nBase, int kBase, int lane) {
  int half = lane >> 4, n = lane & 15;
  const char* base = (const char*)(tile + (nBase + n) * pitch + kBase);
  u32x4 lo = *(const u32x4*)(base + half * 16);
  u32x4 hi = *(const u32x4*)(base + 32 + half * 16);
  return frag_combine(lo, hi);
}

__device__ __forceinline__ f32x8 wmma_bf16(bf16x16 a, bf16x16 b, f32x8 c) {
  return __builtin_amdgcn_wmma_f32_16x16x32_bf16(false, a, false, b, (short)0, c, false, false);
}

__device__ __forceinline__ float waveRedSum(float v) {
#pragma unroll
  for (int o = 16; o; o >>= 1) v += __shfl_xor(v, o, 32);
  return v;
}

// ---------------------------------------------------------------- init

__global__ void init_kernel(float* cmean, float* posbias) {
  int t = blockIdx.x * blockDim.x + threadIdx.x;
  if (t < B_ * H_) cmean[t] = 0.f;
  if (t < S_) {
    float ang = 6.283185307179586f * (float)t / 1024.0f;
    float b = 0.f;
#pragma unroll
    for (int f = 1; f <= 4; f <<= 1) {
      float fa = (float)f * ang;
      b += __sinf(fa) + __cosf(fa);
    }
    posbias[t] = b;
  }
}

__global__ void f32_to_bf16_kernel(const float* __restrict__ in, u16* __restrict__ out, int n) {
  int i = blockIdx.x * blockDim.x + threadIdx.x;
  int stride = gridDim.x * blockDim.x;
  for (; i < n; i += stride) out[i] = f2bf(in[i]);
}

// ---------------------------------------------------------------- GEMM
// C[M][N] = (A[M][K] @ W[N][K]^T + bias[N]) * outScale, bf16 in / bf16 out, f32 accum.
// Block tile 128x128, 8 waves (2 M-groups x 4 N-groups), wave tile 64x32 = 4x2 WMMAs.
// Double-buffered LDS stages filled with async global->LDS copies: stage k+1's
// 4 async ops per thread overlap stage k's 8 WMMAs; ASYNCcnt in-order retirement
// means s_wait_asynccnt<=4 releases the current buffer.

#define GLP 40  // LDS pitch (bf16 elems): 32 + 8 pad

__global__ __launch_bounds__(256) void gemm_bf16_kernel(
    const u16* __restrict__ A, const u16* __restrict__ W,
    const float* __restrict__ bias, u16* __restrict__ C,
    int M, int N, int K, float outScale) {
  __shared__ u16 sA[2][128 * GLP];
  __shared__ u16 sB[2][128 * GLP];
  int tid = threadIdx.x;
  int lane = tid & 31, wid = tid >> 5;
  int wm = wid & 1, wn = wid >> 1;
  int mBase = blockIdx.y * 128, nBase = blockIdx.x * 128;

  f32x8 acc[4][2];
#pragma unroll
  for (int i = 0; i < 4; ++i)
#pragma unroll
    for (int j = 0; j < 2; ++j) acc[i][j] = zero8();

  int r = tid >> 1, ch = tid & 1;
  const u16* gA = A + (size_t)(mBase + r) * K + ch * 16;
  const u16* gB = W + (size_t)(nBase + r) * K + ch * 16;

  auto issue = [&](int buf, int k0) {
    u16* la = &sA[buf][r * GLP + ch * 16];
    u16* lb = &sB[buf][r * GLP + ch * 16];
    async_copy16(gA + k0, la);
    async_copy16(gA + k0 + 8, la + 8);
    async_copy16(gB + k0, lb);
    async_copy16(gB + k0 + 8, lb + 8);
  };

  int nIter = K >> 5;
  issue(0, 0);
  for (int it = 0; it < nIter; ++it) {
    int cur = it & 1;
    if (it + 1 < nIter) {
      __syncthreads();  // all waves done reading buf[cur^1] (iter it-1)
      issue(cur ^ 1, (it + 1) << 5);
      wait_async<4>();  // 4 newest outstanding -> current buffer complete
    } else {
      wait_async<0>();
    }
    __syncthreads();

    bf16x16 af[4];
#pragma unroll
    for (int mt = 0; mt < 4; ++mt)
      af[mt] = a_frag(&sA[cur][(wm * 64 + mt * 16) * GLP], GLP, 0, lane);
    bf16x16 bfr[2];
#pragma unroll
    for (int nt = 0; nt < 2; ++nt)
      bfr[nt] = b_frag(&sB[cur][0], GLP, wn * 32 + nt * 16, 0, lane);
#pragma unroll
    for (int mt = 0; mt < 4; ++mt)
#pragma unroll
      for (int nt = 0; nt < 2; ++nt)
        acc[mt][nt] = wmma_bf16(af[mt], bfr[nt], acc[mt][nt]);
  }

  int half = lane >> 4, nn = lane & 15;
#pragma unroll
  for (int mt = 0; mt < 4; ++mt)
#pragma unroll
    for (int nt = 0; nt < 2; ++nt) {
      int col = nBase + wn * 32 + nt * 16 + nn;
      float bv = bias[col];
      int row0 = mBase + wm * 64 + mt * 16 + half * 8;
#pragma unroll
      for (int rr = 0; rr < 8; ++rr)
        C[(size_t)(row0 + rr) * N + col] = f2bf((acc[mt][nt][rr] + bv) * outScale);
    }
}

// ---------------------------------------------------------------- flash attention
// One workgroup per (query-tile of 16, batch). 8 waves; wave w owns H slice [128w,128w+128).
// Key blocks of 32. Scores reduced across waves via ds_add_f32; wave0 does online softmax.
// Q and K tiles staged with async global->LDS copies; V goes through VGPRs (transposed).

#define QP  1032   // LDS pitch for Q/K tiles (bf16 elems)
#define VTP 40     // LDS pitch for transposed V tile
#define FA_SMEM (16 * QP * 2 + 32 * QP * 2 + 1024 * VTP * 2 + 16 * 32 * 4 + 16 * 32 * 2 + 16 * 4 + 16 * 4)

__global__ __launch_bounds__(256) void flash_attn_kernel(
    const u16* __restrict__ Q, const u16* __restrict__ Kk, const u16* __restrict__ Vv,
    const float* __restrict__ posbias, float* __restrict__ O) {
  extern __shared__ char smem[];
  u16* sQ = (u16*)smem;                 // [16][QP]
  u16* sK = sQ + 16 * QP;               // [32][QP]
  u16* sVt = sK + 32 * QP;              // [1024][VTP]  (Vt[h][key])
  float* sSc = (float*)(sVt + 1024 * VTP);  // [16][32]
  u16* sP = (u16*)(sSc + 16 * 32);          // [16][32]
  float* sAlpha = (float*)(sP + 16 * 32);   // [16]
  float* sL = sAlpha + 16;                  // [16]

  int tid = threadIdx.x, lane = tid & 31, wid = tid >> 5;
  int qBase = blockIdx.x * 16, b = blockIdx.y;
  const size_t bOff = (size_t)b * S_ * H_;

  // stage Q tile [16][1024] asynchronously
  for (int i = tid; i < 16 * 128; i += 256) {
    int rr = i >> 7, c8 = (i & 127) * 8;
    async_copy16(Q + bOff + (size_t)(qBase + rr) * H_ + c8, sQ + rr * QP + c8);
  }

  f32x8 oacc[8];
#pragma unroll
  for (int i = 0; i < 8; ++i) oacc[i] = zero8();

  float m_st = -1e30f, l_st = 0.f;

  int nKB = (qBase + 15) / 32 + 1;
  for (int kb = 0; kb < nKB; ++kb) {
    int t0 = kb * 32;
    __syncthreads();
    // K tile row-major, async
    for (int i = tid; i < 32 * 128; i += 256) {
      int rr = i >> 7, c8 = (i & 127) * 8;
      async_copy16(Kk + bOff + (size_t)(t0 + rr) * H_ + c8, sK + rr * QP + c8);
    }
    // V tile transposed via VGPRs: sVt[h][key]
    for (int i = tid; i < 32 * 128; i += 256) {
      int rr = i >> 7, c8 = (i & 127) * 8;
      u32x4 d = *(const u32x4*)(Vv + bOff + (size_t)(t0 + rr) * H_ + c8);
      const u16* dv = (const u16*)&d;
#pragma unroll
      for (int j = 0; j < 8; ++j) sVt[(c8 + j) * VTP + rr] = dv[j];
    }
    sSc[tid] = 0.f;
    sSc[tid + 256] = 0.f;
    wait_async<0>();
    __syncthreads();

    // partial scores over this wave's 128-wide H slice
    f32x8 sc0 = zero8(), sc1 = zero8();
#pragma unroll
    for (int ks = 0; ks < 4; ++ks) {
      int hb = wid * 128 + ks * 32;
      bf16x16 aq = a_frag(sQ, QP, hb, lane);
      sc0 = wmma_bf16(aq, b_frag(sK, QP, 0, hb, lane), sc0);
      sc1 = wmma_bf16(aq, b_frag(sK, QP, 16, hb, lane), sc1);
    }
    {
      int half = lane >> 4, nn = lane & 15;
#pragma unroll
      for (int rr = 0; rr < 8; ++rr) {
        atomicAdd(&sSc[(rr + half * 8) * 32 + nn], sc0[rr]);
        atomicAdd(&sSc[(rr + half * 8) * 32 + nn + 16], sc1[rr]);
      }
    }
    __syncthreads();

    // wave0: online softmax with positional bias + causal mask
    if (wid == 0 && lane < 16) {
      int row = lane, qAbs = qBase + row;
      float mOld = m_st, mNew = mOld;
      float s[32];
#pragma unroll
      for (int j = 0; j < 32; ++j) {
        float v = sSc[row * 32 + j] + posbias[t0 + j];
        if (t0 + j > qAbs) v = -1e30f;
        s[j] = v;
        mNew = fmaxf(mNew, v);
      }
      float alpha = __expf(mOld - mNew);
      float ls = l_st * alpha;
#pragma unroll
      for (int j = 0; j < 32; ++j) {
        float pv = __expf(s[j] - mNew);
        ls += pv;
        sP[row * 32 + j] = f2bf(pv);
      }
      m_st = mNew; l_st = ls;
      sAlpha[row] = alpha;
      sL[row] = ls;
    }
    __syncthreads();

    // rescale accumulators, then O += P @ V over this wave's slice
    {
      int half = lane >> 4;
      float al[8];
#pragma unroll
      for (int rr = 0; rr < 8; ++rr) al[rr] = sAlpha[rr + half * 8];
#pragma unroll
      for (int nt = 0; nt < 8; ++nt)
#pragma unroll
        for (int rr = 0; rr < 8; ++rr) oacc[nt][rr] *= al[rr];
      bf16x16 ap = a_frag(sP, 32, 0, lane);
#pragma unroll
      for (int nt = 0; nt < 8; ++nt)
        oacc[nt] = wmma_bf16(ap, b_frag(sVt, VTP, wid * 128 + nt * 16, 0, lane), oacc[nt]);
    }
  }
  __syncthreads();
  {
    int half = lane >> 4, nn = lane & 15;
    float linv[8];
#pragma unroll
    for (int rr = 0; rr < 8; ++rr) linv[rr] = 1.0f / sL[rr + half * 8];
#pragma unroll
    for (int nt = 0; nt < 8; ++nt) {
      int col = wid * 128 + nt * 16 + nn;
#pragma unroll
      for (int rr = 0; rr < 8; ++rr) {
        int row = qBase + rr + half * 8;
        O[bOff + (size_t)row * H_ + col] = oacc[nt][rr] * linv[rr];
      }
    }
  }
}

// ---------------------------------------------------------------- routing + mean
// One wave per token. Lane owns h in [32*lane, 32*lane+32) -> groups 2*lane, 2*lane+1.
// Top-16 of 64 group norms via 16 rounds of wave-max + ballot; softmax weights;
// accumulate token-sum into cmean via global atomics (mean division deferred).

__global__ __launch_bounds__(256) void route_mean_kernel(
    const float* __restrict__ attn, float* __restrict__ cmean) {
  int tid = threadIdx.x, lane = tid & 31, wid = tid >> 5;
  int b = blockIdx.x >> 4, chunk = blockIdx.x & 15;
  int s0 = chunk * 128 + wid * 16;

  float acc[32];
#pragma unroll
  for (int i = 0; i < 32; ++i) acc[i] = 0.f;

#pragma unroll 1
  for (int tI = 0; tI < 16; ++tI) {
    int s = s0 + tI;
    const float* base = attn + ((size_t)b * S_ + s) * H_ + lane * 32;
    float v[32];
#pragma unroll
    for (int i = 0; i < 8; ++i) {
      float4 d = ((const float4*)base)[i];
      v[i * 4 + 0] = d.x; v[i * 4 + 1] = d.y; v[i * 4 + 2] = d.z; v[i * 4 + 3] = d.w;
    }
    float n0 = 0.f, n1 = 0.f;
#pragma unroll
    for (int i = 0; i < 16; ++i) { n0 += v[i] * v[i]; n1 += v[16 + i] * v[16 + i]; }
    float r0 = sqrtf(n0), r1 = sqrtf(n1);
    float vmax = 0.f, sum = 0.f, w0 = 0.f, w1 = 0.f;
#pragma unroll 1
    for (int it = 0; it < 16; ++it) {
      float mv = fmaxf(r0, r1);
#pragma unroll
      for (int o = 16; o; o >>= 1) mv = fmaxf(mv, __shfl_xor(mv, o, 32));
      if (it == 0) vmax = mv;
      unsigned long long b0 = __ballot(r0 == mv);
      unsigned long long b1 = __ballot(r1 == mv);
      int l0 = b0 ? (__ffsll(b0) - 1) : 1000;
      int l1 = b1 ? (__ffsll(b1) - 1) : 1000;
      float wv = __expf(mv - vmax);
      sum += wv;
      if (2 * l0 <= 2 * l1 + 1) {
        if (lane == l0) { r0 = -1e30f; w0 = wv; }
      } else {
        if (lane == l1) { r1 = -1e30f; w1 = wv; }
      }
    }
    float inv = 1.0f / sum;
    w0 *= inv; w1 *= inv;
#pragma unroll
    for (int i = 0; i < 16; ++i) { acc[i] += v[i] * w0; acc[16 + i] += v[16 + i] * w1; }
  }
  float* dst = cmean + b * H_ + lane * 32;
#pragma unroll
  for (int i = 0; i < 32; ++i) atomicAdd(&dst[i], acc[i]);
}

// ---------------------------------------------------------------- epilogue projections

// x[b][j] = dot(csum[b,:], Wo[j,:]) / S + bo[j]    (Wo mean folded through the linear map)
__global__ __launch_bounds__(256) void xproj_kernel(
    const float* __restrict__ csum, const float* __restrict__ Wo,
    const float* __restrict__ bo, float* __restrict__ x) {
  int lane = threadIdx.x & 31, wid = threadIdx.x >> 5;
  int j = blockIdx.x * 8 + wid;
  const float* wr = Wo + (size_t)j * H_ + lane * 32;
  float w[32];
#pragma unroll
  for (int i = 0; i < 8; ++i) {
    float4 d = ((const float4*)wr)[i];
    w[i * 4] = d.x; w[i * 4 + 1] = d.y; w[i * 4 + 2] = d.z; w[i * 4 + 3] = d.w;
  }
#pragma unroll 1
  for (int bb = 0; bb < 8; ++bb) {
    const float* cr = csum + bb * H_ + lane * 32;
    float p = 0.f;
#pragma unroll
    for (int i = 0; i < 8; ++i) {
      float4 d = ((const float4*)cr)[i];
      p += d.x * w[i * 4] + d.y * w[i * 4 + 1] + d.z * w[i * 4 + 2] + d.w * w[i * 4 + 3];
    }
    p = waveRedSum(p);
    if (lane == 0) x[bb * D_ + j] = p * (1.0f / 2048.0f) + bo[j];
  }
}

// hidden[b][r] = sum_c dot(x[b, 16c:16c+16], codebook[instr[r][c]])
__global__ __launch_bounds__(256) void decoder_kernel(
    const float* __restrict__ x, const float* __restrict__ codebook,
    const int* __restrict__ instr, float* __restrict__ hidden) {
  int lane = threadIdx.x & 31, wid = threadIdx.x >> 5;
  int rrow = blockIdx.x * 8 + wid;
  int c0 = lane, c1 = lane + 32;
  const float* cb0 = codebook + (size_t)instr[rrow * 64 + c0] * 16;
  const float* cb1 = codebook + (size_t)instr[rrow * 64 + c1] * 16;
  float e0[16], e1[16];
#pragma unroll
  for (int i = 0; i < 16; ++i) { e0[i] = cb0[i]; e1[i] = cb1[i]; }
#pragma unroll 1
  for (int bb = 0; bb < 8; ++bb) {
    const float* xb = x + bb * D_;
    float p = 0.f;
#pragma unroll
    for (int i = 0; i < 16; ++i)
      p += xb[c0 * 16 + i] * e0[i] + xb[c1 * 16 + i] * e1[i];
    p = waveRedSum(p);
    if (lane == 0) hidden[bb * 1024 + rrow] = p;
  }
}

// out[b][v] = dot(hidden[b,:], W_lm[v,:]) + b_lm[v] — bandwidth bound on W_lm (206 MB)
__global__ __launch_bounds__(256) void lm_head_kernel(
    const float* __restrict__ hidden, const float* __restrict__ Wlm,
    const float* __restrict__ blm, float* __restrict__ out) {
  __shared__ float sH[8 * 1024];
  int tid = threadIdx.x;
#pragma unroll
  for (int i = 0; i < 8; ++i) {
    int c4 = (tid + i * 256) * 4;
    async_copy16(hidden + c4, sH + c4);
  }
  wait_async<0>();
  __syncthreads();
  int lane = tid & 31, wid = tid >> 5;
  int v = blockIdx.x * 8 + wid;
  if (v >= V_) return;
  const float* wr = Wlm + (size_t)v * 1024 + lane * 32;
  float w[32];
#pragma unroll
  for (int i = 0; i < 8; ++i) {
    float4 d = ((const float4*)wr)[i];
    w[i * 4] = d.x; w[i * 4 + 1] = d.y; w[i * 4 + 2] = d.z; w[i * 4 + 3] = d.w;
  }
  float bias = blm[v];
#pragma unroll 1
  for (int bb = 0; bb < 8; ++bb) {
    const float* hr = sH + bb * 1024 + lane * 32;
    float p = 0.f;
#pragma unroll
    for (int i = 0; i < 32; ++i) p += hr[i] * w[i];
    p = waveRedSum(p);
    if (lane == 0) out[(size_t)bb * V_ + v] = p + bias;
  }
}

// ---------------------------------------------------------------- launch

extern "C" void kernel_launch(void* const* d_in, const int* in_sizes, int n_in,
                              void* d_out, int out_size, void* d_ws, size_t ws_size,
                              hipStream_t stream) {
  const float* inputs = (const float*)d_in[0];
  const float* Wt = (const float*)d_in[1];
  const float* bt = (const float*)d_in[2];
  const float* Wq = (const float*)d_in[3];
  const float* bq = (const float*)d_in[4];
  const float* Wk = (const float*)d_in[5];
  const float* bk = (const float*)d_in[6];
  const float* Wv = (const float*)d_in[7];
  const float* bv = (const float*)d_in[8];
  const float* Wo = (const float*)d_in[9];
  const float* bo = (const float*)d_in[10];
  const float* codebook = (const float*)d_in[11];
  const int* instr = (const int*)d_in[12];
  const float* Wlm = (const float*)d_in[13];
  const float* blm = (const float*)d_in[14];
  float* out = (float*)d_out;

  char* p = (char*)d_ws;
  auto alloc = [&](size_t bytes) -> char* {
    char* r = p;
    p += (bytes + 255) & ~(size_t)255;
    return r;
  };
  u16* inputs_bf = (u16*)alloc((size_t)B_ * S_ * D_ * 2);
  u16* Wt_bf = (u16*)alloc((size_t)H_ * D_ * 2);
  u16* Wq_bf = (u16*)alloc((size_t)H_ * H_ * 2);
  u16* Wk_bf = (u16*)alloc((size_t)H_ * H_ * 2);
  u16* Wv_bf = (u16*)alloc((size_t)H_ * H_ * 2);
  u16* tok_bf = (u16*)alloc((size_t)B_ * S_ * H_ * 2);
  u16* q_bf = (u16*)alloc((size_t)B_ * S_ * H_ * 2);
  u16* k_bf = (u16*)alloc((size_t)B_ * S_ * H_ * 2);
  u16* v_bf = (u16*)alloc((size_t)B_ * S_ * H_ * 2);
  float* attn = (float*)alloc((size_t)B_ * S_ * H_ * 4);
  float* posbias = (float*)alloc((size_t)S_ * 4);
  float* cmean = (float*)alloc((size_t)B_ * H_ * 4);
  float* x = (float*)alloc((size_t)B_ * D_ * 4);
  float* hidden = (float*)alloc((size_t)B_ * 1024 * 4);

  init_kernel<<<32, 256, 0, stream>>>(cmean, posbias);
  f32_to_bf16_kernel<<<2048, 256, 0, stream>>>(inputs, inputs_bf, B_ * S_ * D_);
  f32_to_bf16_kernel<<<512, 256, 0, stream>>>(Wt, Wt_bf, H_ * D_);
  f32_to_bf16_kernel<<<512, 256, 0, stream>>>(Wq, Wq_bf, H_ * H_);
  f32_to_bf16_kernel<<<512, 256, 0, stream>>>(Wk, Wk_bf, H_ * H_);
  f32_to_bf16_kernel<<<512, 256, 0, stream>>>(Wv, Wv_bf, H_ * H_);

  const int M = B_ * S_;
  gemm_bf16_kernel<<<dim3(H_ / 128, M / 128), 256, 0, stream>>>(
      inputs_bf, Wt_bf, bt, tok_bf, M, H_, D_, 1.0f);
  gemm_bf16_kernel<<<dim3(H_ / 128, M / 128), 256, 0, stream>>>(
      tok_bf, Wq_bf, bq, q_bf, M, H_, H_, 0.03125f);  // 1/sqrt(H) folded into q
  gemm_bf16_kernel<<<dim3(H_ / 128, M / 128), 256, 0, stream>>>(
      tok_bf, Wk_bf, bk, k_bf, M, H_, H_, 1.0f);
  gemm_bf16_kernel<<<dim3(H_ / 128, M / 128), 256, 0, stream>>>(
      tok_bf, Wv_bf, bv, v_bf, M, H_, H_, 1.0f);

  (void)hipFuncSetAttribute((const void*)flash_attn_kernel,
                            hipFuncAttributeMaxDynamicSharedMemorySize, FA_SMEM);
  flash_attn_kernel<<<dim3(S_ / 16, B_), 256, FA_SMEM, stream>>>(
      q_bf, k_bf, v_bf, posbias, attn);

  route_mean_kernel<<<B_ * 16, 256, 0, stream>>>(attn, cmean);
  xproj_kernel<<<D_ / 8, 256, 0, stream>>>(cmean, Wo, bo, x);
  decoder_kernel<<<1024 / 8, 256, 0, stream>>>(x, codebook, instr, hidden);
  lm_head_kernel<<<(V_ + 7) / 8, 256, 0, stream>>>(hidden, Wlm, blm, out);
}
